// B_Biformer_SR_54030688583933
// MI455X (gfx1250) — matrix operand order repaired and checked by reading, not compiled, not run on hardware
//
#include <hip/hip_runtime.h>
#include <hip/hip_bf16.h>
#include <math.h>

typedef __attribute__((ext_vector_type(16))) _Float16 v16h;
typedef __attribute__((ext_vector_type(8)))  _Float16 v8h;
typedef __attribute__((ext_vector_type(8)))  float    v8f;

#define NWIN 4

// -------------------------------------------------------------------------
// WMMA GEMM, 2x2 register-blocked: each wave computes a 32x32 supertile of
// C[M,N] = A[M,K]f16 * B[N,K]f16^T + bias (+epilogue, +residual).
// 4 v_wmma_f32_16x16x32_f16 per 32-wide K step; A/B fragments reused 2x.
// Requires M,N multiples of 16 (odd 16-tile counts handled by clamping).
// epi: 0 = none, 1 = exact GELU, 2 = sigmoid
// -------------------------------------------------------------------------
__global__ void k_gemm(const _Float16* __restrict__ A, const _Float16* __restrict__ B,
                       const float* __restrict__ bias, float* __restrict__ C,
                       const float* __restrict__ res,
                       int M, int N, int K, int lda, int ldb, int ldc, int ldres,
                       int sA, int sB, int sC, int epi)
{
  const int lane = threadIdx.x & 31;
  const int wave = threadIdx.x >> 5;
  const int ntm2 = (M + 31) >> 5;
  const int ntn2 = (N + 31) >> 5;
  const int st   = blockIdx.x * (blockDim.x >> 5) + wave;
  if (st >= ntm2 * ntn2) return;               // wave-uniform: EXEC stays all-ones
  const int sm = st / ntn2;
  const int sn = st - sm * ntn2;
  const int r  = lane & 15;
  const int g  = lane >> 4;
  const int z  = blockIdx.z;

  const int row0 = sm * 32, col0 = sn * 32;
  const bool hasM1 = (row0 + 16) < M;          // second row-subtile valid?
  const bool hasN1 = (col0 + 16) < N;          // second col-subtile valid?

  const _Float16* Az = A + (size_t)z * sA;
  const _Float16* Bz = B + (size_t)z * sB;
  const _Float16* A0 = Az + (size_t)(row0 + r) * lda;
  const _Float16* A1 = Az + (size_t)(row0 + (hasM1 ? 16 : 0) + r) * lda;
  const _Float16* B0 = Bz + (size_t)(col0 + r) * ldb;
  const _Float16* B1 = Bz + (size_t)(col0 + (hasN1 ? 16 : 0) + r) * ldb;

  v8f acc00 = {}, acc01 = {}, acc10 = {}, acc11 = {};
  for (int k = 0; k < K; k += 32) {
    union { v16h v; v8h h[2]; } a0, a1;
    // ISA 16-bit A 16x32 layout: lanes 0-15 hold K 0..7/16..23, lanes 16-31 hold 8..15/24..31
    a0.h[0] = *(const v8h*)(A0 + k + g * 8);
    a0.h[1] = *(const v8h*)(A0 + k + 16 + g * 8);
    a1.h[0] = *(const v8h*)(A1 + k + g * 8);
    a1.h[1] = *(const v8h*)(A1 + k + 16 + g * 8);
    // ISA 16-bit B 32x16 layout: lanes 0-15 hold K 0..7/16..23, lanes 16-31 hold 8..15/24..31 of column r
    union { v16h v; v8h h[2]; } bu0, bu1;
    bu0.h[0] = *(const v8h*)(B0 + k + g * 8);
    bu0.h[1] = *(const v8h*)(B0 + k + 16 + g * 8);
    bu1.h[0] = *(const v8h*)(B1 + k + g * 8);
    bu1.h[1] = *(const v8h*)(B1 + k + 16 + g * 8);
    v16h b0 = bu0.v;
    v16h b1 = bu1.v;
    if (k + 32 < K) {
      __builtin_prefetch(A0 + k + 32, 0, 3);   // global_prefetch_b8
      __builtin_prefetch(A1 + k + 32, 0, 3);
      __builtin_prefetch(B0 + k + 32, 0, 3);
      __builtin_prefetch(B1 + k + 32, 0, 3);
    }
    acc00 = __builtin_amdgcn_wmma_f32_16x16x32_f16(false, a0.v, false, b0, (short)0, acc00, false, false);
    acc01 = __builtin_amdgcn_wmma_f32_16x16x32_f16(false, a0.v, false, b1, (short)0, acc01, false, false);
    acc10 = __builtin_amdgcn_wmma_f32_16x16x32_f16(false, a1.v, false, b0, (short)0, acc10, false, false);
    acc11 = __builtin_amdgcn_wmma_f32_16x16x32_f16(false, a1.v, false, b1, (short)0, acc11, false, false);
  }

  float* Cz = C + (size_t)z * sC;
  const bool hasres = (res != nullptr);
  auto emit = [&](v8f acc, int rb, int cb) {
    const int col = cb + r;
    const float bc = bias ? bias[col] : 0.0f;
#pragma unroll
    for (int v = 0; v < 8; ++v) {
      int row = rb + g * 8 + v;                // ISA f32 C/D layout
      float val = acc[v] + bc;
      if (epi == 1)      val = 0.5f * val * (1.0f + erff(val * 0.70710678118f));
      else if (epi == 2) val = 1.0f / (1.0f + __expf(-val));
      if (hasres) val += res[(size_t)row * ldres + col];
      Cz[(size_t)row * ldc + col] = val;
    }
  };
  emit(acc00, row0, col0);
  if (hasN1) emit(acc01, row0, col0 + 16);
  if (hasM1) emit(acc10, row0 + 16, col0);
  if (hasM1 && hasN1) emit(acc11, row0 + 16, col0 + 16);
}

// ------------------------- elementwise / helper kernels -------------------
__global__ void k_pack_f16(const float* __restrict__ src, _Float16* __restrict__ dst,
                           int rows, int K, int Kpad, int ldsrc, int rows_src)
{
  int i = blockIdx.x * 256 + threadIdx.x;
  int total = rows * Kpad;
  if (i >= total) return;
  int rr = i / Kpad, j = i - rr * Kpad;
  float v = (rr < rows_src && j < K) ? src[(size_t)rr * ldsrc + j] : 0.0f;
  dst[i] = (_Float16)v;
}

__global__ void k_padcopy(const float* __restrict__ src, float* __restrict__ dst, int n, int npad)
{
  int i = blockIdx.x * 256 + threadIdx.x;
  if (i >= npad) return;
  dst[i] = (i < n) ? src[i] : 0.0f;
}

__global__ void k_copy(float* __restrict__ dst, const float* __restrict__ src, int n)
{
  int i = blockIdx.x * 256 + threadIdx.x;
  if (i < n) dst[i] = src[i];
}

__global__ void k_add(float* __restrict__ dst, const float* __restrict__ src, int n)
{
  int i = blockIdx.x * 256 + threadIdx.x;
  if (i < n) dst[i] += src[i];
}

__global__ void k_ln(const float* __restrict__ x, float* __restrict__ y,
                     const float* __restrict__ g, const float* __restrict__ b,
                     int rows, int C, int ldx, int ldy)
{
  int rr = blockIdx.x * 256 + threadIdx.x;
  if (rr >= rows) return;
  const float* xr = x + (size_t)rr * ldx;
  float mu = 0.0f;
  for (int cc = 0; cc < C; ++cc) mu += xr[cc];
  mu /= C;
  float var = 0.0f;
  for (int cc = 0; cc < C; ++cc) { float d = xr[cc] - mu; var += d * d; }
  var /= C;
  float inv = rsqrtf(var + 1e-6f);
  float* yr = y + (size_t)rr * ldy;
  for (int cc = 0; cc < C; ++cc)
    yr[cc] = (xr[cc] - mu) * inv * g[cc] + b[cc];
}

__global__ void k_prep(const float* __restrict__ x, float* __restrict__ y) // NCHW[3,96,96]->[HW,3], -mean
{
  int i = blockIdx.x * 256 + threadIdx.x;
  const int HW = 96 * 96;
  if (i >= HW * 3) return;
  int pix = i / 3, cc = i - pix * 3;
  float mean = (cc == 0) ? 0.4488f : (cc == 1) ? 0.4371f : 0.404f;
  y[i] = x[cc * HW + pix] - mean;
}

__global__ void k_im2col(const float* __restrict__ src, _Float16* __restrict__ dst,
                         int h, int w, int cin, int ldsrc, int Kpad, int stride, int ho, int wo)
{
  int i = blockIdx.x * 256 + threadIdx.x;
  int total = ho * wo * Kpad;
  if (i >= total) return;
  int j = i % Kpad;
  int pix = i / Kpad;
  int oy = pix / wo, ox = pix - oy * wo;
  float v = 0.0f;
  int K = cin * 9;
  if (j < K) {
    int cc = j / 9, t = j - cc * 9, ky = t / 3, kx = t - ky * 3;
    int iy = oy * stride + ky - 1, ix = ox * stride + kx - 1;
    if (iy >= 0 && iy < h && ix >= 0 && ix < w)
      v = src[((size_t)iy * w + ix) * ldsrc + cc];
  }
  dst[i] = (_Float16)v;
}

__device__ __forceinline__ int winrow(int p, int t, int h, int w)
{
  int wh = h / NWIN, ww = w / NWIN;
  int wy = p / NWIN, wx = p - wy * NWIN;
  int iy = t / ww, ix = t - iy * ww;
  return (wy * wh + iy) * w + (wx * ww + ix);
}

__global__ void k_lepe(const float* __restrict__ v, const float* __restrict__ w5,
                       const float* __restrict__ b, float* __restrict__ out,
                       int h, int w, int C, int ldv)
{
  int i = blockIdx.x * 256 + threadIdx.x;
  int total = h * w * C;
  if (i >= total) return;
  int cc = i % C;
  int pix = i / C;
  int oy = pix / w, ox = pix - oy * w;
  float s = b[cc];
  for (int ky = 0; ky < 5; ++ky) {
    int iy = oy + ky - 2;
    if (iy < 0 || iy >= h) continue;
    for (int kx = 0; kx < 5; ++kx) {
      int ix = ox + kx - 2;
      if (ix < 0 || ix >= w) continue;
      s += w5[cc * 25 + ky * 5 + kx] * v[((size_t)iy * w + ix) * ldv + cc];
    }
  }
  out[(size_t)pix * C + cc] = s;
}

__global__ void k_win_mean(const float* __restrict__ x, float* __restrict__ out,
                           int ld, int C, int h, int w)
{
  int i = blockIdx.x * 256 + threadIdx.x;
  if (i >= 16 * C) return;
  int p = i / C, cc = i - p * C;
  int wh = h / NWIN, ww = w / NWIN;
  int wy = p / NWIN, wx = p - wy * NWIN;
  float s = 0.0f;
  for (int iy = 0; iy < wh; ++iy)
    for (int ix = 0; ix < ww; ++ix)
      s += x[((size_t)(wy * wh + iy) * w + wx * ww + ix) * ld + cc];
  out[i] = s / (float)(wh * ww);
}

__global__ void k_route(const float* __restrict__ qwin, const float* __restrict__ kwin,
                        int C, int topk, float scale, int* __restrict__ idx)
{
  int p = threadIdx.x;
  if (p >= 16) return;
  float logit[16];
  for (int q2 = 0; q2 < 16; ++q2) {
    float s = 0.0f;
    for (int cc = 0; cc < C; ++cc) s += qwin[p * C + cc] * kwin[q2 * C + cc];
    logit[q2] = s * scale;
  }
  for (int t = 0; t < topk; ++t) {
    int best = 0; float bv = -3.4e38f;
    for (int q2 = 0; q2 < 16; ++q2)
      if (logit[q2] > bv) { bv = logit[q2]; best = q2; }
    idx[p * topk + t] = best;
    logit[best] = -3.4e38f;
  }
}

__global__ void k_pack_q(const float* __restrict__ qkv, _Float16* __restrict__ qf,
                         int p, int h, int w, int C, int ldqkv, int nh, float scale)
{
  int hd = C / nh;
  int L = (h / NWIN) * (w / NWIN);
  int i = blockIdx.x * 256 + threadIdx.x;
  if (i >= nh * L * 32) return;
  int d = i & 31;
  int mi = i >> 5;
  int t = mi % L, m = mi / L;
  float v = 0.0f;
  if (d < hd) {
    int row = winrow(p, t, h, w);
    v = qkv[(size_t)row * ldqkv + m * hd + d] * scale;
  }
  qf[i] = (_Float16)v;
}

__global__ void k_gather_k(const float* __restrict__ qkv, const int* __restrict__ idx,
                           _Float16* __restrict__ kf, int p, int h, int w, int C, int ldqkv,
                           int nh, int topk)
{
  int hd = C / nh;
  int L = (h / NWIN) * (w / NWIN);
  int keys = topk * L;
  int i = blockIdx.x * 256 + threadIdx.x;
  if (i >= nh * keys * 32) return;
  int d = i & 31;
  int mj = i >> 5;
  int j = mj % keys, m = mj / keys;
  float v = 0.0f;
  if (d < hd) {
    int pw = idx[p * topk + j / L];
    int row = winrow(pw, j % L, h, w);
    v = qkv[(size_t)row * ldqkv + C + m * hd + d];
  }
  kf[i] = (_Float16)v;
}

__global__ void k_gather_vT(const float* __restrict__ qkv, const int* __restrict__ idx,
                            _Float16* __restrict__ vf, int p, int h, int w, int C, int ldqkv,
                            int nh, int topk)
{
  int hd = C / nh;
  int L = (h / NWIN) * (w / NWIN);
  int keys = topk * L;
  int i = blockIdx.x * 256 + threadIdx.x;
  if (i >= nh * 32 * keys) return;
  int j = i % keys;
  int md = i / keys;
  int d = md & 31, m = md >> 5;
  float v = 0.0f;
  if (d < hd) {
    int pw = idx[p * topk + j / L];
    int row = winrow(pw, j % L, h, w);
    v = qkv[(size_t)row * ldqkv + 2 * C + m * hd + d];
  }
  vf[i] = (_Float16)v;
}

__global__ void k_softmax(float* __restrict__ x, int rows, int n)
{
  int rr = blockIdx.x * 256 + threadIdx.x;
  if (rr >= rows) return;
  float* xr = x + (size_t)rr * n;
  float mx = -3.4e38f;
  for (int j = 0; j < n; ++j) mx = fmaxf(mx, xr[j]);
  float s = 0.0f;
  for (int j = 0; j < n; ++j) { float e = __expf(xr[j] - mx); xr[j] = e; s += e; }
  float inv = 1.0f / s;
  for (int j = 0; j < n; ++j) xr[j] *= inv;
}

__global__ void k_scatter(const float* __restrict__ attout, float* __restrict__ out,
                          int p, int h, int w, int C, int nh)
{
  int hd = C / nh;
  int L = (h / NWIN) * (w / NWIN);
  int i = blockIdx.x * 256 + threadIdx.x;
  if (i >= nh * L * hd) return;
  int d = i % hd;
  int mt = i / hd;
  int t = mt % L, m = mt / L;
  int row = winrow(p, t, h, w);
  out[(size_t)row * C + m * hd + d] = attout[((size_t)m * L + t) * 32 + d];
}

__global__ void k_cmean(const float* __restrict__ x, float* __restrict__ out,
                        int rows, int C, int ld)
{
  int cc = blockIdx.x * 256 + threadIdx.x;
  if (cc >= C) return;
  float s = 0.0f;
  for (int rr = 0; rr < rows; ++rr) s += x[(size_t)rr * ld + cc];
  out[cc] = s / (float)rows;
}

__global__ void k_fc_small(const float* __restrict__ in, const float* __restrict__ w,
                           const float* __restrict__ b, float* __restrict__ out,
                           int cout, int cin, int act)
{
  int o = threadIdx.x;
  if (o >= cout) return;
  float s = b[o];
  for (int i = 0; i < cin; ++i) s += w[o * cin + i] * in[i];
  if (act == 1) s = fmaxf(s, 0.0f);
  else if (act == 2) s = 1.0f / (1.0f + __expf(-s));
  out[o] = s;
}

__global__ void k_scse(float* __restrict__ x, const float* __restrict__ cvec,
                       const float* __restrict__ smap, int rows, int C, int ld)
{
  int i = blockIdx.x * 256 + threadIdx.x;
  if (i >= rows * C) return;
  int cc = i % C;
  int rr = i / C;
  size_t o = (size_t)rr * ld + cc;
  x[o] = x[o] * cvec[cc] + x[o] * smap[o];
}

__global__ void k_upsample2(const float* __restrict__ in, float* __restrict__ out,
                            int h, int w, int C)
{
  int i = blockIdx.x * 256 + threadIdx.x;
  int W2 = w * 2;
  if (i >= 4 * h * w * C) return;
  int cc = i % C;
  int pix = i / C;
  int oy = pix / W2, ox = pix - oy * W2;
  out[i] = in[((size_t)(oy >> 1) * w + (ox >> 1)) * C + cc];
}

__global__ void k_concat(const float* __restrict__ a, int lda, int Ca,
                         const float* __restrict__ b, int ldb, int Cb,
                         float* __restrict__ out, int rows)
{
  int Co = Ca + Cb;
  int i = blockIdx.x * 256 + threadIdx.x;
  if (i >= rows * Co) return;
  int cc = i % Co;
  int rr = i / Co;
  out[i] = (cc < Ca) ? a[(size_t)rr * lda + cc] : b[(size_t)rr * ldb + (cc - Ca)];
}

__global__ void k_shuffle_out(const float* __restrict__ in, float* __restrict__ out)
{
  int i = blockIdx.x * 256 + threadIdx.x;
  const int HO = 192, WO = 192;
  if (i >= 3 * HO * WO) return;
  int cc = i / (HO * WO);
  int rem = i - cc * (HO * WO);
  int oy = rem / WO, ox = rem - oy * WO;
  int y = oy >> 1, x = ox >> 1, r1 = oy & 1, r2 = ox & 1;
  float mean = (cc == 0) ? 0.4488f : (cc == 1) ? 0.4371f : 0.404f;
  out[i] = in[((size_t)y * 96 + x) * 16 + cc * 4 + r1 * 2 + r2] + mean;
}

// ------------------------------ host side --------------------------------
struct Ctx {
  hipStream_t s;
  void* const* din;
  float *xbuf, *skip, *tln, *qkv, *tmp2, *attn, *lepe, *smap, *scores, *attout,
        *xup, *xcat, *biasp, *qwin, *kwin, *pooled, *z1, *z2;
  int* idx;
  _Float16 *a16, *w16, *qf, *kf, *vf, *att16;
};

#define PF(i) ((const float*)c.din[i])

// pytree (sorted-key) flat parameter indices; d_in[0] = x
enum {
  CONCAT_B = 1, CONCAT_W = 2,
  DOWNC_B  = 3, DOWNC_W  = 4,
  DOWNLN_B = 5, DOWNLN_G = 6,
  L_DOWN   = 7,
  L_UP     = 43,
  L_MID0   = 79,
  NORM_B   = 259, NORM_G   = 260,
  NORMUP_B = 261, NORMUP_G = 262,
  PATCHC_B = 263, PATCHC_W = 264,
  PATCHLN_B = 265, PATCHLN_G = 266,
  RECON_B  = 267, RECON_W  = 268,
  UPC_B    = 269, UPC_W    = 270,
  UPLN_B   = 271, UPLN_G   = 272
};
// block base B: lepe b/w +0/1, qkv b/w +2/3, wo b/w +4/5, fc1 b/w +6/7,
//               fc2 b/w +8/9, ln1 b/g +10/11, ln2 b/g +12/13
// layer base L: blocks at L+14*d, conv b/w L+28/29, scse c1 L+30/31, c2 L+32/33, s L+34/35

static inline int pad16(int x) { return (x + 15) & ~15; }
static inline int pad32(int x) { return (x + 31) & ~31; }
static inline int nblk(int n)  { return (n + 255) / 256; }

static void gemm(Ctx& c, const _Float16* A, const _Float16* B, const float* bias,
                 float* C, const float* res, int M, int N, int K,
                 int lda, int ldb, int ldc, int ldres,
                 int batch, int sA, int sB, int sC, int epi)
{
  int tiles = ((M + 31) / 32) * ((N + 31) / 32);
  dim3 grid((tiles + 7) / 8, 1, batch);
  k_gemm<<<grid, 256, 0, c.s>>>(A, B, bias, C, res, M, N, K, lda, ldb, ldc, ldres, sA, sB, sC, epi);
}

// out[M,Np] = src[M,:K] @ W^T + bias (+res), W at din[wi] as [N,K]
static void linear(Ctx& c, const float* src, int srcld, int M, int K, int N,
                   int wi, int bi, float* out, int ldout, const float* res, int ldres, int epi)
{
  int Kp = pad32(K), Np = pad16(N);
  k_pack_f16<<<nblk(M * Kp), 256, 0, c.s>>>(src, c.a16, M, K, Kp, srcld, M);
  k_pack_f16<<<nblk(Np * Kp), 256, 0, c.s>>>(PF(wi), c.w16, Np, K, Kp, K, N);
  const float* bp = PF(bi);
  if (Np != N) { k_padcopy<<<nblk(Np), 256, 0, c.s>>>(PF(bi), c.biasp, N, Np); bp = c.biasp; }
  gemm(c, c.a16, c.w16, bp, out, res, M, Np, Kp, Kp, Kp, ldout, ldres, 1, 0, 0, 0, epi);
}

static void conv3(Ctx& c, const float* src, int srcld, int h, int w, int cin,
                  int stride, int cout, int wi, int bi, float* out, int ldout, int epi)
{
  int ho = h / stride, wo = w / stride;
  int K = cin * 9, Kp = pad32(K), Np = pad16(cout), M = ho * wo;
  k_im2col<<<nblk(M * Kp), 256, 0, c.s>>>(src, c.a16, h, w, cin, srcld, Kp, stride, ho, wo);
  k_pack_f16<<<nblk(Np * Kp), 256, 0, c.s>>>(PF(wi), c.w16, Np, K, Kp, K, cout);
  const float* bp = PF(bi);
  if (Np != cout) { k_padcopy<<<nblk(Np), 256, 0, c.s>>>(PF(bi), c.biasp, cout, Np); bp = c.biasp; }
  gemm(c, c.a16, c.w16, bp, out, nullptr, M, Np, Kp, Kp, Kp, ldout, 0, 1, 0, 0, 0, epi);
}

static void run_block(Ctx& c, float* x, int xld, int h, int w, int C, int nh, int topk, int B0)
{
  int M = h * w, C3 = 3 * C;
  int C3p = pad16(C3);                         // qkv buffer leading dim (192 / 368)
  float scale = 1.0f / sqrtf((float)C);
  // ln1
  k_ln<<<nblk(M), 256, 0, c.s>>>(x, c.tln, PF(B0 + 11), PF(B0 + 10), M, C, xld, C);
  // qkv
  linear(c, c.tln, C, M, C, C3, B0 + 3, B0 + 2, c.qkv, C3p, nullptr, 0, 0);
  // lepe (depthwise 5x5 on v)
  k_lepe<<<nblk(M * C), 256, 0, c.s>>>(c.qkv + 2 * C, PF(B0 + 1), PF(B0 + 0), c.lepe, h, w, C, C3p);
  // routing
  k_win_mean<<<nblk(16 * C), 256, 0, c.s>>>(c.qkv, c.qwin, C3p, C, h, w);
  k_win_mean<<<nblk(16 * C), 256, 0, c.s>>>(c.qkv + C, c.kwin, C3p, C, h, w);
  k_route<<<1, 32, 0, c.s>>>(c.qwin, c.kwin, C, topk, scale, c.idx);
  int L = (h / NWIN) * (w / NWIN);
  int keys = topk * L;
  for (int p = 0; p < 16; ++p) {
    k_pack_q<<<nblk(nh * L * 32), 256, 0, c.s>>>(c.qkv, c.qf, p, h, w, C, C3p, nh, scale);
    k_gather_k<<<nblk(nh * keys * 32), 256, 0, c.s>>>(c.qkv, c.idx, c.kf, p, h, w, C, C3p, nh, topk);
    k_gather_vT<<<nblk(nh * 32 * keys), 256, 0, c.s>>>(c.qkv, c.idx, c.vf, p, h, w, C, C3p, nh, topk);
    // scores[nh, L, keys] = q @ k^T
    gemm(c, c.qf, c.kf, nullptr, c.scores, nullptr, L, keys, 32, 32, 32, keys, 0,
         nh, L * 32, keys * 32, L * keys, 0);
    k_softmax<<<nblk(nh * L), 256, 0, c.s>>>(c.scores, nh * L, keys);
    k_pack_f16<<<nblk(nh * L * keys), 256, 0, c.s>>>(c.scores, c.att16, nh * L, keys, keys, keys, nh * L);
    // out[nh, L, 32] = att @ v
    gemm(c, c.att16, c.vf, nullptr, c.attout, nullptr, L, 32, keys, keys, keys, 32, 0,
         nh, L * keys, 32 * keys, L * 32, 0);
    k_scatter<<<nblk(nh * L * (C / nh)), 256, 0, c.s>>>(c.attout, c.attn, p, h, w, C, nh);
  }
  k_add<<<nblk(M * C), 256, 0, c.s>>>(c.attn, c.lepe, M * C);
  // wo + residual
  linear(c, c.attn, C, M, C, C, B0 + 5, B0 + 4, x, xld, x, xld, 0);
  // MLP
  k_ln<<<nblk(M), 256, 0, c.s>>>(x, c.tln, PF(B0 + 13), PF(B0 + 12), M, C, xld, C);
  int hid = 2 * C;
  linear(c, c.tln, C, M, C, hid, B0 + 7, B0 + 6, c.tmp2, hid, nullptr, 0, 1 /*gelu*/);
  linear(c, c.tmp2, hid, M, hid, C, B0 + 9, B0 + 8, x, xld, x, xld, 0);
}

static void run_layer(Ctx& c, float* x, int xld, int h, int w, int C, int nh, int topk, int L0)
{
  for (int d = 0; d < 2; ++d) run_block(c, x, xld, h, w, C, nh, topk, L0 + 14 * d);
  int M = h * w;
  // scse
  k_cmean<<<nblk(C), 256, 0, c.s>>>(x, c.pooled, M, C, xld);
  int cr = C / 16;
  k_fc_small<<<1, 64, 0, c.s>>>(c.pooled, PF(L0 + 31), PF(L0 + 30), c.z1, cr, C, 1);
  k_fc_small<<<1, 256, 0, c.s>>>(c.z1, PF(L0 + 33), PF(L0 + 32), c.z2, C, cr, 2);
  linear(c, x, xld, M, C, C, L0 + 35, L0 + 34, c.smap, pad16(C), nullptr, 0, 2 /*sigmoid*/);
  k_scse<<<nblk(M * C), 256, 0, c.s>>>(x, c.z2, c.smap, M, C, xld);
  // 3x3 conv C->C (im2col snapshot first, so writing into x is safe)
  conv3(c, x, xld, h, w, C, 1, C, L0 + 29, L0 + 28, x, xld, 0);
}

extern "C" void kernel_launch(void* const* d_in, const int* in_sizes, int n_in,
                              void* d_out, int out_size, void* d_ws, size_t ws_size,
                              hipStream_t stream)
{
  (void)in_sizes; (void)n_in; (void)out_size; (void)ws_size;
  Ctx c;
  c.s = stream;
  c.din = d_in;

  char* base = (char*)d_ws;
  size_t off = 0;
  auto alloc = [&](size_t bytes) -> void* {
    void* r = base + off;
    off += (bytes + 255) & ~(size_t)255;
    return r;
  };
  const int HWf = 96 * 96;
  const int HWm = 48 * 48;
  (void)HWm;
  c.xbuf   = (float*)alloc((size_t)HWf * 128 * 4);
  c.skip   = (float*)alloc((size_t)HWf * 64 * 4);
  c.tln    = (float*)alloc((size_t)HWf * 128 * 4);
  c.qkv    = (float*)alloc((size_t)HWf * 192 * 4);   // covers 9216x192 and 2304x368
  c.tmp2   = (float*)alloc((size_t)HWf * 256 * 4);
  c.attn   = (float*)alloc((size_t)HWf * 128 * 4);
  c.lepe   = (float*)alloc((size_t)HWf * 128 * 4);
  c.smap   = (float*)alloc((size_t)HWf * 128 * 4);
  c.scores = (float*)alloc((size_t)1000000 * 4);
  c.attout = (float*)alloc((size_t)60000 * 4);
  c.xup    = (float*)alloc((size_t)HWf * 64 * 4);
  c.xcat   = (float*)alloc((size_t)HWf * 120 * 4);
  c.biasp  = (float*)alloc(512 * 4);
  c.qwin   = (float*)alloc(16 * 128 * 4);
  c.kwin   = (float*)alloc(16 * 128 * 4);
  c.pooled = (float*)alloc(128 * 4);
  c.z1     = (float*)alloc(64 * 4);
  c.z2     = (float*)alloc(128 * 4);
  c.idx    = (int*)alloc(16 * 8 * 4);
  c.a16    = (_Float16*)alloc((size_t)HWf * 1088 * 2);
  c.w16    = (_Float16*)alloc((size_t)131072 * 2);
  c.qf     = (_Float16*)alloc((size_t)60000 * 2);
  c.kf     = (_Float16*)alloc((size_t)230000 * 2);
  c.vf     = (_Float16*)alloc((size_t)230000 * 2);
  c.att16  = (_Float16*)alloc((size_t)1000000 * 2);

  // ---- shallow feature extraction ----
  k_prep<<<nblk(HWf * 3), 256, 0, stream>>>((const float*)d_in[0], c.tmp2);
  conv3(c, c.tmp2, 3, 96, 96, 3, 1, 60, PATCHC_W, PATCHC_B, c.xbuf, 64, 0);
  k_ln<<<nblk(HWf), 256, 0, stream>>>(c.xbuf, c.xbuf, PF(PATCHLN_G), PF(PATCHLN_B), HWf, 60, 64, 64);
  k_copy<<<nblk(HWf * 64), 256, 0, stream>>>(c.skip, c.xbuf, HWf * 64);

  // ---- encoder ----
  run_layer(c, c.xbuf, 64, 96, 96, 60, 3, 1, L_DOWN);
  k_ln<<<nblk(HWf), 256, 0, stream>>>(c.xbuf, c.tln, PF(DOWNLN_G), PF(DOWNLN_B), HWf, 60, 64, 64);
  conv3(c, c.tln, 64, 96, 96, 60, 2, 120, DOWNC_W, DOWNC_B, c.xbuf, 120, 0);
  k_ln<<<nblk(HWm), 256, 0, stream>>>(c.xbuf, c.xbuf, PF(NORM_G), PF(NORM_B), HWm, 120, 120, 120);

  // ---- mid ----
  for (int i = 0; i < 5; ++i)
    run_layer(c, c.xbuf, 120, 48, 48, 120, 6, 8, L_MID0 + 36 * i);
  k_ln<<<nblk(HWm), 256, 0, stream>>>(c.xbuf, c.xbuf, PF(NORM_G), PF(NORM_B), HWm, 120, 120, 120);
  k_ln<<<nblk(HWm), 256, 0, stream>>>(c.xbuf, c.tln, PF(UPLN_G), PF(UPLN_B), HWm, 120, 120, 120);

  // ---- decoder ----
  k_upsample2<<<nblk(HWf * 120), 256, 0, stream>>>(c.tln, c.tmp2, 48, 48, 120);
  conv3(c, c.tmp2, 120, 96, 96, 120, 1, 60, UPC_W, UPC_B, c.xup, 64, 0);
  k_concat<<<nblk(HWf * 120), 256, 0, stream>>>(c.xup, 64, 60, c.skip, 64, 60, c.xcat, HWf);
  conv3(c, c.xcat, 120, 96, 96, 120, 1, 60, CONCAT_W, CONCAT_B, c.xbuf, 64, 0);
  run_layer(c, c.xbuf, 64, 96, 96, 60, 3, 1, L_UP);
  k_ln<<<nblk(HWf), 256, 0, stream>>>(c.xbuf, c.xbuf, PF(NORMUP_G), PF(NORMUP_B), HWf, 60, 64, 64);

  // ---- reconstruction ----
  conv3(c, c.xbuf, 64, 96, 96, 60, 1, 12, RECON_W, RECON_B, c.tmp2, 16, 0);
  k_shuffle_out<<<nblk(3 * 192 * 192), 256, 0, stream>>>(c.tmp2, (float*)d_out);
}
